// GNetVerOne_50620484551320
// MI455X (gfx1250) — matrix operand order repaired and checked by reading, not compiled, not run on hardware
//
#include <hip/hip_runtime.h>

#define BATCH 8
#define SEQ   2048
#define HIN   20
#define KNN1  16
#define KNN2  32
#define NG_   32
#define DM_   128
#define DH_   32
#define ROWS  (BATCH*SEQ)   // 16384
#define KDIM  640           // NG_*HIN, K of GEMM1 (= 20 steps of 32)

typedef unsigned int uint;
typedef __attribute__((ext_vector_type(4)))  uint     u32x4;
typedef __attribute__((ext_vector_type(16))) _Float16 v16h;
typedef __attribute__((ext_vector_type(8)))  float    v8f;

union FragH { v16h h; u32x4 q[2]; };

__device__ __forceinline__ uint pack_h2(float a, float b) {
  union { _Float16 h[2]; uint u; } p;
  p.h[0] = (_Float16)a; p.h[1] = (_Float16)b;
  return p.u;
}

// ---------------------------------------------------------------------------
// K0: pre-swizzle op_kernel (640x128) and emb_W (128x32) into WMMA B-fragment
// order: Bp[kstep][ntile][lane][reg] so each lane's 8 regs are 32 contiguous B.
// B layout (16-bit 32x16): lanes 0-15 K=0..7,16..23 (2/VGPR), lanes 16-31 K=8..15,24..31.
__global__ void k_pack(const float* __restrict__ opk, const float* __restrict__ embW,
                       uint* __restrict__ Bp1, uint* __restrict__ Bp2) {
  int t = blockIdx.x * blockDim.x + threadIdx.x;
  if (t < 20*8*32*8) {            // GEMM1: 20 ksteps x 8 ntiles
    int j = t & 7, lane = (t >> 3) & 31, nt = (t >> 8) & 7, ks = t >> 11;
    int n  = nt*16 + (lane & 15);
    int k0 = ks*32 + ((lane >> 4) * 8) + 2*j + ((j >> 2) * 8);
    Bp1[t] = pack_h2(opk[(size_t)k0*DM_ + n], opk[(size_t)(k0+1)*DM_ + n]);
  }
  if (t < 4*2*32*8) {             // GEMM2: 4 ksteps x 2 ntiles
    int j = t & 7, lane = (t >> 3) & 31, nt = (t >> 8) & 1, ks = t >> 9;
    int n  = nt*16 + (lane & 15);
    int k0 = ks*32 + ((lane >> 4) * 8) + 2*j + ((j >> 2) * 8);
    Bp2[t] = pack_h2(embW[(size_t)k0*DH_ + n], embW[(size_t)(k0+1)*DH_ + n]);
  }
}

// ---------------------------------------------------------------------------
// K1: top-32 nearest neighbors per row. Coords staged in LDS (broadcast reads),
// per-thread sorted u64 list (d2-bits<<32 | idx) in LDS, tie-break lower index.
__global__ void __launch_bounds__(128) k_nn(const float* __restrict__ frame,
                                            int* __restrict__ idx2) {
  __shared__ float sc[SEQ*3];                 // 24 KB
  __shared__ unsigned long long lst[32*128];  // 32 KB
  int tid = threadIdx.x;
  int b   = blockIdx.x >> 4;                  // 16 blocks per batch
  int lq  = ((blockIdx.x & 15) << 7) + tid;
  for (int i = tid; i < SEQ*3; i += 128) {
    int l = i / 3, d = i - l*3;
    sc[i] = frame[(size_t)(b*SEQ + l)*12 + d];
  }
  __syncthreads();
  float qx = sc[lq*3+0], qy = sc[lq*3+1], qz = sc[lq*3+2];
  float sq_q = qx*qx + qy*qy + qz*qz;
  for (int s = 0; s < 32; ++s) lst[s*128 + tid] = ~0ull;
  unsigned long long worst = ~0ull;
  for (int j = 0; j < SEQ; ++j) {
    float cx = sc[j*3], cy = sc[j*3+1], cz = sc[j*3+2];
    // identical formula to reference (sq_i + sq_j - 2*dot) to match ordering
    float d2 = sq_q + (cx*cx + cy*cy + cz*cz) - 2.f*(qx*cx + qy*cy + qz*cz);
    unsigned u = __float_as_uint(d2);
    u = (u & 0x80000000u) ? ~u : (u | 0x80000000u);   // total order on floats
    unsigned long long key = ((unsigned long long)u << 32) | (unsigned)j;
    if (key < worst) {
      int p = 31;
      while (p > 0 && lst[(p-1)*128 + tid] > key) {
        lst[p*128 + tid] = lst[(p-1)*128 + tid]; --p;
      }
      lst[p*128 + tid] = key;
      worst = lst[31*128 + tid];
    }
  }
  size_t base = (size_t)(b*SEQ + lq) * 32;
  for (int s = 0; s < 32; ++s) idx2[base + s] = (int)(lst[s*128 + tid] & 0xffffffffu);
}

// ---------------------------------------------------------------------------
// K2: per row, compute 7-d geometric features + 32 gaussians over K1=16
// neighbors and emit the fp16 A row: A[row, g*20+h] = sum_k g1[k,g]*attr[j_k,h]
// One wave per row; lane = gaussian index (keeps 49-entry precision matrix in
// registers per lane, coalesced constant loads).
__global__ void __launch_bounds__(128) k_buildA(
    const float* __restrict__ frame, const int* __restrict__ seq,
    const float* __restrict__ attr,  const float* __restrict__ c1,
    const float* __restrict__ p1,    const int* __restrict__ idx2,
    _Float16* __restrict__ A) {
  __shared__ float sAttr[4][KNN1][HIN];
  int tid = threadIdx.x, w = tid >> 5, lane = tid & 31;
  int row = blockIdx.x*4 + w;
  int b   = row >> 11;
  const float* Fl = frame + (size_t)row*12;
  float clx=Fl[0], cly=Fl[1], clz=Fl[2];
  float a0x=Fl[3], a0y=Fl[4], a0z=Fl[5];
  float a1x=Fl[6], a1y=Fl[7], a1z=Fl[8];
  float a2x=Fl[9], a2y=Fl[10], a2z=Fl[11];   // axis2 == z_l
  float sl = (float)seq[row];
  float ctr[7], P[7][7];
  for (int d = 0; d < 7; ++d) ctr[d] = c1[d*32 + lane];
  for (int d = 0; d < 7; ++d)
    for (int k = 0; k < 7; ++k) P[d][k] = p1[(d*7 + k)*32 + lane];
  float g1k[KNN1];
  for (int k = 0; k < KNN1; ++k) {
    int j = idx2[(size_t)row*32 + k];
    const float* Fj = frame + ((size_t)(b*SEQ) + j)*12;
    float dx = Fj[0]-clx, dy = Fj[1]-cly, dzc = Fj[2]-clz;
    float zjx = Fj[9], zjy = Fj[10], zjz = Fj[11];
    float dist = sqrtf(dx*dx + dy*dy + dzc*dzc + 1e-12f);
    float x[7];
    x[0] = dx*a0x + dy*a0y + dzc*a0z;
    x[1] = dx*a1x + dy*a1y + dzc*a1z;
    x[2] = dx*a2x + dy*a2y + dzc*a2z;
    x[3] = fminf(fabsf((float)seq[b*SEQ + j] - sl), 8.f);
    x[4] = a2x*zjx + a2y*zjy + a2z*zjz;         // z.z_j
    x[5] = (dx*zjx + dy*zjy + dzc*zjz) / dist;  // delta.z_j / d
    x[6] = (a2x*dx + a2y*dy + a2z*dzc) / dist;  // z.delta / d
    float diff[7];
    for (int d = 0; d < 7; ++d) diff[d] = x[d] - ctr[d];
    float s = 0.f;
    for (int kk = 0; kk < 7; ++kk) {
      float y = 0.f;
      for (int d = 0; d < 7; ++d) y += diff[d]*P[d][kk];
      s += y*y;
    }
    g1k[k] = __expf(-0.5f*s);
    if (lane < HIN) sAttr[w][k][lane] = attr[((size_t)(b*SEQ) + j)*HIN + lane];
  }
  __syncthreads();
  _Float16* Arow = A + (size_t)row*KDIM + lane*HIN;   // lane == gaussian g
  for (int h = 0; h < HIN; ++h) {
    float acc = 0.f;
    for (int k = 0; k < KNN1; ++k) acc += g1k[k]*sAttr[w][k][h];
    Arow[h] = (_Float16)acc;
  }
}

// ---------------------------------------------------------------------------
// K3: fused WMMA GEMMs. Wave computes a 32x128 panel (two 16-row M-tiles that
// share every B fragment -> 2 WMMAs per B load). All 8 B fragments of a K-step
// are loaded up front: LOADcnt decrements in order, so the scheduler can use
// partial s_wait_loadcnt values and keep later fragments in flight while the
// first pairs of WMMAs execute. Then f is re-fragmented through LDS for
// emb = relu(f @ W2[128x32] + b2).
__global__ void __launch_bounds__(128) k_gemm(
    const _Float16* __restrict__ A, const uint* __restrict__ Bp1,
    const uint* __restrict__ Bp2,   const float* __restrict__ bias1,
    const float* __restrict__ bias2, float* __restrict__ emb) {
  __shared__ _Float16 ft[4*32*DM_];   // 32 KB: per-wave 32x128 fp16 f-tile
  int tid = threadIdx.x, w = tid >> 5, lane = tid & 31;
  int half = lane >> 4, nlo = lane & 15;
  int mBase = blockIdx.x*128 + w*32;
  v8f acc0[8], acc1[8];
#pragma unroll
  for (int i = 0; i < 8; ++i)
    for (int v = 0; v < 8; ++v) { acc0[i][v] = 0.f; acc1[i][v] = 0.f; }
  const _Float16* aPtr0 = A + (size_t)(mBase + nlo)*KDIM + half*8;
  const _Float16* aPtr1 = aPtr0 + (size_t)16*KDIM;
  const u32x4* bBase = reinterpret_cast<const u32x4*>(Bp1);  // frag = 2 u32x4
  for (int ks = 0; ks < 20; ++ks) {
    FragH a0, a1;
    const u32x4* pa0 = reinterpret_cast<const u32x4*>(aPtr0 + ks*32);
    const u32x4* pa1 = reinterpret_cast<const u32x4*>(aPtr1 + ks*32);
    a0.q[0] = pa0[0]; a0.q[1] = pa0[2];       // K {0..7,16..23}+8*half
    a1.q[0] = pa1[0]; a1.q[1] = pa1[2];
    if (ks < 19) {                            // stream-prefetch next A chunks
      __builtin_prefetch(aPtr0 + (ks+1)*32, 0, 0);
      __builtin_prefetch(aPtr1 + (ks+1)*32, 0, 0);
    }
    FragH b[8];                               // whole K-step B panel in flight
#pragma unroll
    for (int nt = 0; nt < 8; ++nt) {
      const u32x4* pb = bBase + ((ks*8 + nt)*32 + lane)*2;
      b[nt].q[0] = pb[0]; b[nt].q[1] = pb[1];
    }
#pragma unroll
    for (int nt = 0; nt < 8; ++nt) {
      acc0[nt] = __builtin_amdgcn_wmma_f32_16x16x32_f16(
          false, a0.h, false, b[nt].h, (short)0, acc0[nt], false, false);
      acc1[nt] = __builtin_amdgcn_wmma_f32_16x16x32_f16(
          false, a1.h, false, b[nt].h, (short)0, acc1[nt], false, false);
    }
  }
  // bias + relu, park fp16 f-tiles in LDS (C layout -> row-major)
  _Float16* myft = ft + w*32*DM_;
#pragma unroll
  for (int nt = 0; nt < 8; ++nt) {
    float bv = bias1[nt*16 + nlo];
#pragma unroll
    for (int v = 0; v < 8; ++v) {
      myft[(v + half*8)*DM_      + nt*16 + nlo] = (_Float16)fmaxf(acc0[nt][v] + bv, 0.f);
      myft[(16 + v + half*8)*DM_ + nt*16 + nlo] = (_Float16)fmaxf(acc1[nt][v] + bv, 0.f);
    }
  }
  __syncthreads();
  v8f e0[2], e1[2];
#pragma unroll
  for (int i = 0; i < 2; ++i)
    for (int v = 0; v < 8; ++v) { e0[i][v] = 0.f; e1[i][v] = 0.f; }
  for (int ks = 0; ks < 4; ++ks) {
    FragH a0, a1;
    const u32x4* pa0 = reinterpret_cast<const u32x4*>(myft + nlo*DM_      + ks*32 + half*8);
    const u32x4* pa1 = reinterpret_cast<const u32x4*>(myft + (16+nlo)*DM_ + ks*32 + half*8);
    a0.q[0] = pa0[0]; a0.q[1] = pa0[2];
    a1.q[0] = pa1[0]; a1.q[1] = pa1[2];
#pragma unroll
    for (int nt = 0; nt < 2; ++nt) {
      FragH bf;
      const u32x4* pb = reinterpret_cast<const u32x4*>(Bp2) + ((ks*2 + nt)*32 + lane)*2;
      bf.q[0] = pb[0]; bf.q[1] = pb[1];
      e0[nt] = __builtin_amdgcn_wmma_f32_16x16x32_f16(
          false, a0.h, false, bf.h, (short)0, e0[nt], false, false);
      e1[nt] = __builtin_amdgcn_wmma_f32_16x16x32_f16(
          false, a1.h, false, bf.h, (short)0, e1[nt], false, false);
    }
  }
#pragma unroll
  for (int nt = 0; nt < 2; ++nt) {
    float bv = bias2[nt*16 + nlo];
#pragma unroll
    for (int v = 0; v < 8; ++v) {
      emb[(size_t)(mBase + v + half*8)*DH_      + nt*16 + nlo] = fmaxf(e0[nt][v] + bv, 0.f);
      emb[(size_t)(mBase + 16 + v + half*8)*DH_ + nt*16 + nlo] = fmaxf(e1[nt][v] + bv, 0.f);
    }
  }
}

// ---------------------------------------------------------------------------
// K4: the five tiny heads from emb (32-dim dots), one thread per row.
__global__ void k_heads(const float* __restrict__ emb,
    const float* bW, const float* bb, const float* sW, const float* sb,
    const float* cW, const float* cb, const float* nW, const float* nb,
    float* __restrict__ heads) {
  int t = blockIdx.x*blockDim.x + threadIdx.x;
  if (t >= ROWS) return;
  const float* e = emb + (size_t)t*DH_;
  float hb = 0.f, hs = 0.f, hc = 0.f, n0 = 0.f, n1 = 0.f;
  for (int i = 0; i < DH_; ++i) {
    float v = e[i];
    hb += v*bW[i]; hs += v*sW[i]; hc += v*cW[i];
    n0 += v*nW[i*2]; n1 += v*nW[i*2 + 1];
  }
  float* H = heads + (size_t)t*5;
  H[0] = fmaxf(hb + bb[0], 0.f);
  H[1] = fmaxf(hs + sb[0], 0.f);
  H[2] = fmaxf(hc + cb[0], 0.f);
  H[3] = fmaxf(n0 + nb[0], 0.f);
  H[4] = fmaxf(n1 + nb[1], 0.f);
}

// ---------------------------------------------------------------------------
// K5: final attention. One wave per row, lane = neighbor k (K2=32 == wave32):
// features -> 32 gaussians -> gw, then wave-wide max/sum via shfl_xor.
__global__ void __launch_bounds__(128) k_attn(
    const float* __restrict__ frame, const int* __restrict__ seq,
    const int* __restrict__ idx2,    const float* __restrict__ heads,
    const float* __restrict__ c2,    const float* __restrict__ p2,
    const float* __restrict__ e2W,   const float* __restrict__ e2b,
    float* __restrict__ out) {
  __shared__ float sc2[5*32];
  __shared__ float sp2[25*32];
  __shared__ float sw2[32];
  int tid = threadIdx.x;
  for (int i = tid; i < 160; i += 128) sc2[i] = c2[i];
  for (int i = tid; i < 800; i += 128) sp2[i] = p2[i];
  if (tid < 32) sw2[tid] = e2W[tid];
  __syncthreads();
  int w = tid >> 5, lane = tid & 31;
  int row = blockIdx.x*4 + w;
  int b   = row >> 11;
  const float* Fl = frame + (size_t)row*12;
  float clx=Fl[0], cly=Fl[1], clz=Fl[2];
  float zlx=Fl[9], zly=Fl[10], zlz=Fl[11];
  float sl   = (float)seq[row];
  float beta = heads[(size_t)row*5 + 0];
  float satt = heads[(size_t)row*5 + 1];
  int j = idx2[(size_t)row*32 + lane];
  const float* Fj = frame + ((size_t)(b*SEQ) + j)*12;
  float dx = Fj[0]-clx, dy = Fj[1]-cly, dzc = Fj[2]-clz;
  float zjx = Fj[9], zjy = Fj[10], zjz = Fj[11];
  float dist = sqrtf(dx*dx + dy*dy + dzc*dzc + 1e-12f);
  float x[5];
  x[0] = dist;
  x[1] = zlx*zjx + zly*zjy + zlz*zjz;
  x[2] = (dx*zjx + dy*zjy + dzc*zjz) / dist;
  x[3] = (zlx*dx + zly*dy + zlz*dzc) / dist;
  x[4] = fminf(fabsf((float)seq[b*SEQ + j] - sl), 8.f);
  float gwacc = e2b[0];
  for (int n = 0; n < 32; ++n) {
    float diff[5];
    for (int d = 0; d < 5; ++d) diff[d] = x[d] - sc2[d*32 + n];
    float s = 0.f;
    for (int kk = 0; kk < 5; ++kk) {
      float y = 0.f;
      for (int d = 0; d < 5; ++d) y += diff[d]*sp2[(d*5 + kk)*32 + n];
      s += y*y;
    }
    gwacc += __expf(-0.5f*s) * sw2[n];
  }
  float gw = fmaxf(gwacc, 0.f);
  float e  = (lane == 0) ? satt : heads[((size_t)(b*SEQ) + j)*5 + 2];
  float logit = beta * e;
  float m = logit;
  for (int o = 16; o; o >>= 1) m = fmaxf(m, __shfl_xor(m, o, 32));
  float wgt = gw * __expf(logit - m);
  float n0 = heads[((size_t)(b*SEQ) + j)*5 + 3];
  float n1 = heads[((size_t)(b*SEQ) + j)*5 + 4];
  float S = wgt, o0 = wgt*n0, o1 = wgt*n1;
  for (int o = 16; o; o >>= 1) {
    S  += __shfl_xor(S,  o, 32);
    o0 += __shfl_xor(o0, o, 32);
    o1 += __shfl_xor(o1, o, 32);
  }
  if (lane == 0) {
    float inv = 1.f / (S + 1e-6f);
    out[(size_t)row*2 + 0] = o0*inv;
    out[(size_t)row*2 + 1] = o1*inv;
  }
}

// ---------------------------------------------------------------------------
extern "C" void kernel_launch(void* const* d_in, const int* in_sizes, int n_in,
                              void* d_out, int out_size, void* d_ws, size_t ws_size,
                              hipStream_t stream) {
  const float* attr  = (const float*)d_in[0];
  const float* frame = (const float*)d_in[1];
  const int*   seq   = (const int*)  d_in[2];
  const float* c1    = (const float*)d_in[3];
  const float* p1    = (const float*)d_in[4];
  const float* opk   = (const float*)d_in[5];
  const float* opb   = (const float*)d_in[6];
  const float* embW  = (const float*)d_in[7];
  const float* embB  = (const float*)d_in[8];
  const float* bW    = (const float*)d_in[9];
  const float* bb    = (const float*)d_in[10];
  const float* sW    = (const float*)d_in[11];
  const float* sb    = (const float*)d_in[12];
  const float* cW    = (const float*)d_in[13];
  const float* cb    = (const float*)d_in[14];
  const float* nW    = (const float*)d_in[15];
  const float* nb    = (const float*)d_in[16];
  const float* c2    = (const float*)d_in[17];
  const float* p2    = (const float*)d_in[18];
  const float* e2W   = (const float*)d_in[19];
  const float* e2b   = (const float*)d_in[20];
  (void)in_sizes; (void)n_in; (void)out_size; (void)ws_size;

  // workspace layout (~25.7 MB)
  char* ws = (char*)d_ws;
  int*      idx2  = (int*)ws;                          //  2,097,152 B
  _Float16* Amat  = (_Float16*)(ws + 2097152u);        // 20,971,520 B
  uint*     Bp1   = (uint*)(ws + 23068672u);           //    163,840 B
  uint*     Bp2   = (uint*)(ws + 23232512u);           //      8,192 B
  float*    emb   = (float*)(ws + 23240704u);          //  2,097,152 B
  float*    heads = (float*)(ws + 25337856u);          //    327,680 B
  float*    out   = (float*)d_out;

  k_pack  <<<dim3(160),  dim3(256), 0, stream>>>(opk, embW, Bp1, Bp2);
  k_nn    <<<dim3(128),  dim3(128), 0, stream>>>(frame, idx2);
  k_buildA<<<dim3(4096), dim3(128), 0, stream>>>(frame, seq, attr, c1, p1, idx2, Amat);
  k_gemm  <<<dim3(128),  dim3(128), 0, stream>>>(Amat, Bp1, Bp2, opb, embB, emb);
  k_heads <<<dim3(64),   dim3(256), 0, stream>>>(emb, bW, bb, sW, sb, cW, cb, nW, nb, heads);
  k_attn  <<<dim3(4096), dim3(128), 0, stream>>>(frame, seq, idx2, heads, c2, p2, e2W, e2b, out);
}
